// AR_Transcriber_40518721470918
// MI455X (gfx1250) — matrix operand order repaired and checked by reading, not compiled, not run on hardware
//
#include <hip/hip_runtime.h>

typedef _Float16 f16;
typedef __attribute__((ext_vector_type(16))) _Float16 v16h;
typedef __attribute__((ext_vector_type(8)))  _Float16 v8h;
typedef __attribute__((ext_vector_type(8)))  float    v8f;

#define DEV static __device__ __forceinline__

// ---- problem constants (from reference) ----
constexpr int Bv = 8, Tn = 512, Fm = 229;
constexpr int C1 = 48, C3v = 96;
constexpr int F2 = 114, F4 = 57;
constexpr int Hh = 768, G4 = 3072, K1 = 944, FCK = 5472;  // FCK = 96*57
constexpr int NP = 88, NC = 5, NPOST = 440, PREVN = 176;
constexpr int NBLK = 48;   // persistent LSTM kernel: 48 blocks x 8 waves = 384 waves
// padded activation geometry (1-element spatial halo, zero-filled => no bounds checks)
constexpr int P1T = Tn + 2, P1R = Fm + 2;   // x1 padded: [B][514][231][48]
constexpr int P2T = Tn + 2, P2R = F2 + 2;   // x2p padded: [B][514][116][48]

DEV float sigmf(float x) { return 1.0f / (1.0f + __expf(-x)); }

DEV v8h ld8(const f16* p) { return *(const v8h*)p; }          // 16B global_load_b128
DEV v8h zero8() { v8h z = {}; return z; }
DEV v16h cat16(v8h lo, v8h hi) {
  return __builtin_shufflevector(lo, hi, 0, 1, 2, 3, 4, 5, 6, 7,
                                 8, 9, 10, 11, 12, 13, 14, 15);
}

// ============================================================
// 16x16 WMMA tile over K range [kbeg,kend), runtime loop, double-buffered.
// A supplied as 8-element contiguous groups by a_grp(kg) (unconditional loads;
// zero-row trick handles invalid rows). W row-major, leading dim Kld.
// ============================================================
template <typename GF>
DEV v8f gemm_rng(v8f acc, GF&& a_grp, const f16* __restrict__ W, long Kld,
                 int n0, int Nmax, int kbeg, int kend) {
  const int lane = threadIdx.x & 31;
  const int half = lane >> 4;
  const int lm   = lane & 15;
  const int n    = n0 + lm;
  const bool nok = (n < Nmax);
  const f16* Wr  = W + (long)n * Kld;
  auto ldA0 = [&](int k0) { int g = k0 + 8 * half;      return (g < kend) ? a_grp(g) : zero8(); };
  auto ldA1 = [&](int k0) { int g = k0 + 8 * half + 16; return (g < kend) ? a_grp(g) : zero8(); };
  auto ldB0 = [&](int k0) { int g = k0 + 8 * half;      return (nok && g < kend) ? ld8(Wr + g) : zero8(); };
  auto ldB1 = [&](int k0) { int g = k0 + 8 * half + 16; return (nok && g < kend) ? ld8(Wr + g) : zero8(); };
  v8h a0 = ldA0(kbeg), a1 = ldA1(kbeg), b0 = ldB0(kbeg), b1 = ldB1(kbeg);
  for (int k0 = kbeg; k0 < kend; k0 += 32) {
    int k1 = k0 + 32;
    v8h na0 = zero8(), na1 = zero8(), nb0 = zero8(), nb1 = zero8();
    if (k1 < kend) { na0 = ldA0(k1); na1 = ldA1(k1); nb0 = ldB0(k1); nb1 = ldB1(k1); }
    acc = __builtin_amdgcn_wmma_f32_16x16x32_f16(false, cat16(a0, a1),
                                                 false, cat16(b0, b1),
                                                 (short)0, acc, false, false);
    a0 = na0; a1 = na1; b0 = nb0; b1 = nb1;
  }
  return acc;
}

// compile-time-K variant: fully unrolled, kg constant-folds into load offsets
template <int K, typename GF>
DEV v8f gemm_ct(v8f acc, GF&& a_grp, const f16* __restrict__ W, int n0, int Nmax) {
  const int lane = threadIdx.x & 31;
  const int half = lane >> 4;
  const int lm   = lane & 15;
  const int n    = n0 + lm;
  const bool nok = (n < Nmax);
  const f16* Wr  = W + (long)n * K;
#pragma unroll
  for (int k0 = 0; k0 < K; k0 += 32) {
    int ga = k0 + 8 * half;
    int gb = ga + 16;
    v8h a0 = (ga < K) ? a_grp(ga) : zero8();
    v8h a1 = (gb < K) ? a_grp(gb) : zero8();
    int gw = k0 + 8 * half;
    v8h b0 = (nok && gw < K)      ? ld8(Wr + gw)      : zero8();
    v8h b1 = (nok && gw + 16 < K) ? ld8(Wr + gw + 16) : zero8();
    acc = __builtin_amdgcn_wmma_f32_16x16x32_f16(false, cat16(a0, a1),
                                                 false, cat16(b0, b1),
                                                 (short)0, acc, false, false);
  }
  return acc;
}

// ============================================================
// utility / conversion kernels
// ============================================================
__global__ void k_zero8(f16* __restrict__ p, int ngrp) {   // zero n*8 f16, vectorized
  int i = blockIdx.x * blockDim.x + threadIdx.x;
  if (i < ngrp) *(v8h*)(p + (long)i * 8) = zero8();
}

__global__ void k_f32_to_f16(const float* __restrict__ s, f16* __restrict__ d, int n) {
  int i = blockIdx.x * blockDim.x + threadIdx.x;
  if (i < n) d[i] = (f16)s[i];
}

// conv weights (O,I,3,3) -> f16 [O][r*I + ci]   (K axis reordered to r-major)
__global__ void k_cvt_conv(const float* __restrict__ s, f16* __restrict__ d,
                           int O, int I) {
  int i = blockIdx.x * blockDim.x + threadIdx.x;
  int n = O * I * 9;
  if (i >= n) return;
  int o = i / (9 * I); int rem = i - o * 9 * I;
  int r = rem / I; int ci = rem - r * I;
  d[i] = (f16)s[(o * I + ci) * 9 + r];
}

// fc weights [N][c*57+f] -> f16 [N][f*96+c]   (K axis matches channels-last x3p)
__global__ void k_cvt_fc(const float* __restrict__ s, f16* __restrict__ d, int n) {
  int i = blockIdx.x * blockDim.x + threadIdx.x;
  if (i >= n) return;
  int nn = i / FCK; int kk = i - nn * FCK;
  int f = kk / C3v; int c = kk - f * C3v;
  d[i] = (f16)s[nn * FCK + c * F4 + f];
}

// ============================================================
// conv1 (1ch in) + BN + ReLU -> padded channels-last x1p interior
// ============================================================
__global__ void k_conv1_cl(const float* __restrict__ mel, const float* __restrict__ w,
                           const float* __restrict__ cb, const float* __restrict__ g,
                           const float* __restrict__ bb, const float* __restrict__ mm,
                           const float* __restrict__ vv, f16* __restrict__ x1p, int n) {
  int i = blockIdx.x * blockDim.x + threadIdx.x;
  if (i >= n) return;
  int cg = i % (C1 / 8); int rest = i / (C1 / 8);
  int f = rest % Fm; int bt = rest / Fm;
  int t = bt & (Tn - 1); int b = bt >> 9;
  float s[8];
#pragma unroll
  for (int j = 0; j < 8; ++j) s[j] = 0.0f;
#pragma unroll
  for (int kh = 0; kh < 3; ++kh) {
    int tt = t + kh - 1;
    if ((unsigned)tt >= (unsigned)Tn) continue;
#pragma unroll
    for (int kw = 0; kw < 3; ++kw) {
      int ff = f + kw - 1;
      if ((unsigned)ff >= (unsigned)Fm) continue;
      float mv = mel[((b << 9) + tt) * Fm + ff];
#pragma unroll
      for (int j = 0; j < 8; ++j)
        s[j] += mv * w[(cg * 8 + j) * 9 + kh * 3 + kw];
    }
  }
  v8h o;
#pragma unroll
  for (int j = 0; j < 8; ++j) {
    int co = cg * 8 + j;
    float scale = g[co] * rsqrtf(vv[co] + 1e-5f);
    float y = (s[j] + cb[co] - mm[co]) * scale + bb[co];
    o[j] = (f16)fmaxf(y, 0.0f);
  }
  *(v8h*)(x1p + (((long)b * P1T + t + 1) * P1R + f + 1) * C1 + cg * 8) = o;
}

// ============================================================
// conv2 implicit GEMM (M=B*T*229, K=9*48, N=48) + BN/ReLU
// reads padded x1p (no bounds checks), writes unpadded y2
// ============================================================
__global__ void k_conv2_wmma(const f16* __restrict__ x1p, const f16* __restrict__ W,
                             const float* __restrict__ cb, const float* __restrict__ g,
                             const float* __restrict__ bb, const float* __restrict__ mm,
                             const float* __restrict__ vv, f16* __restrict__ y2) {
  const int MT = (Bv * Tn * Fm) / 16;   // 58624
  int wtile = (blockIdx.x * blockDim.x + threadIdx.x) >> 5;
  if (wtile >= MT * 3) return;
  int ntile = wtile % 3, mtile = wtile / 3;
  int lane = threadIdx.x & 31, half = lane >> 4, lm = lane & 15;
  int m = mtile * 16 + lm;
  int f = m % Fm; int bt = m / Fm; int t = bt & (Tn - 1); int b = bt >> 9;
  // halo-padded base at (t,f): neighbor (dt,df) = base + ((r/3)*P1R + r%3)*48
  const f16* base = x1p + (((long)b * P1T + t) * P1R + f) * C1;
  auto a_grp = [&](int kg) -> v8h {          // kg compile-time after unroll
    int r = kg / C1; int ci0 = kg - r * C1;
    return ld8(base + ((r / 3) * P1R + (r % 3)) * C1 + ci0);
  };
  v8f acc = {};
  acc = gemm_ct<C1 * 9>(acc, a_grp, W, ntile * 16, C1);
  int n = ntile * 16 + lm;                     // co
  float scale = g[n] * rsqrtf(vv[n] + 1e-5f);
  float shift = bb[n] + (cb[n] - mm[n]) * scale;
#pragma unroll
  for (int r = 0; r < 8; ++r) {
    int ms = mtile * 16 + r + 8 * half;
    int ff = ms % Fm; int btt = ms / Fm;
    float y = fmaxf(acc[r] * scale + shift, 0.0f);
    y2[((long)btt * Fm + ff) * C1 + n] = (f16)y;
  }
}

// ============================================================
// conv3 implicit GEMM (M=B*T*114, K=9*48, N=96) + BN/ReLU
// reads padded x2pp, writes unpadded y3
// ============================================================
__global__ void k_conv3_wmma(const f16* __restrict__ x2pp, const f16* __restrict__ W,
                             const float* __restrict__ cb, const float* __restrict__ g,
                             const float* __restrict__ bb, const float* __restrict__ mm,
                             const float* __restrict__ vv, f16* __restrict__ y3) {
  const int MT = (Bv * Tn * F2) / 16;   // 29184
  int wtile = (blockIdx.x * blockDim.x + threadIdx.x) >> 5;
  if (wtile >= MT * 6) return;
  int ntile = wtile % 6, mtile = wtile / 6;
  int lane = threadIdx.x & 31, half = lane >> 4, lm = lane & 15;
  int m = mtile * 16 + lm;
  int f = m % F2; int bt = m / F2; int t = bt & (Tn - 1); int b = bt >> 9;
  const f16* base = x2pp + (((long)b * P2T + t) * P2R + f) * C1;
  auto a_grp = [&](int kg) -> v8h {
    int r = kg / C1; int ci0 = kg - r * C1;
    return ld8(base + ((r / 3) * P2R + (r % 3)) * C1 + ci0);
  };
  v8f acc = {};
  acc = gemm_ct<C1 * 9>(acc, a_grp, W, ntile * 16, C3v);
  int n = ntile * 16 + lm;
  float scale = g[n] * rsqrtf(vv[n] + 1e-5f);
  float shift = bb[n] + (cb[n] - mm[n]) * scale;
#pragma unroll
  for (int r = 0; r < 8; ++r) {
    int ms = mtile * 16 + r + 8 * half;
    int ff = ms % F2; int btt = ms / F2;
    float y = fmaxf(acc[r] * scale + shift, 0.0f);
    y3[((long)btt * F2 + ff) * C3v + n] = (f16)y;
  }
}

// ============================================================
// pooling (channels-last, 8 ch per thread)
// ============================================================
// y2 (unpadded 229) -> x2pp interior (padded 116 row)
__global__ void k_pool_pad(const f16* __restrict__ in, f16* __restrict__ out, int ngrp) {
  int i = blockIdx.x * blockDim.x + threadIdx.x;
  if (i >= ngrp) return;
  int cg = i % (C1 / 8); int rest = i / (C1 / 8);
  int f = rest % F2; int bt = rest / F2;
  int t = bt & (Tn - 1); int b = bt >> 9;
  const f16* p0 = in + ((long)bt * Fm + 2 * f) * C1 + cg * 8;
  v8h a = ld8(p0), bq = ld8(p0 + C1), o;
#pragma unroll
  for (int j = 0; j < 8; ++j) o[j] = (f16)fmaxf((float)a[j], (float)bq[j]);
  *(v8h*)(out + (((long)b * P2T + t + 1) * P2R + f + 1) * C1 + cg * 8) = o;
}

// y3 (unpadded 114) -> x3p (unpadded 57)
__global__ void k_pool_cl(const f16* __restrict__ in, f16* __restrict__ out,
                          int ngrp, int Fin, int Fout, int C) {
  int i = blockIdx.x * blockDim.x + threadIdx.x;
  if (i >= ngrp) return;
  int cg = i % (C / 8); int rest = i / (C / 8);
  int f = rest % Fout; int bt = rest / Fout;
  const f16* p0 = in + ((long)bt * Fin + 2 * f) * C + cg * 8;
  v8h a = ld8(p0), b = ld8(p0 + C), o;
#pragma unroll
  for (int j = 0; j < 8; ++j) o[j] = (f16)fmaxf((float)a[j], (float)b[j]);
  *(v8h*)(out + ((long)bt * Fout + f) * C + cg * 8) = o;
}

// ============================================================
// FC: acoustic[B*T,768] = x3p-rows (contiguous K=5472) @ Wfc'^T + b  (WMMA)
// ============================================================
__global__ void k_fc_wmma(const f16* __restrict__ x3p, const f16* __restrict__ W,
                          const float* __restrict__ fcb, f16* __restrict__ ac) {
  const int MT = (Bv * Tn) / 16;   // 256
  int wtile = (blockIdx.x * blockDim.x + threadIdx.x) >> 5;
  if (wtile >= MT * 48) return;
  int ntile = wtile % 48, mtile = wtile / 48;
  int lane = threadIdx.x & 31, half = lane >> 4, lm = lane & 15;
  int m = mtile * 16 + lm;        // m = b*512 + t
  const f16* arow = x3p + (long)m * FCK;        // channels-last row is contiguous
  auto a_grp = [&](int kg) -> v8h { return ld8(arow + kg); };
  v8f acc = {};
  acc = gemm_rng(acc, a_grp, W, FCK, ntile * 16, Hh, 0, FCK);
  int n = ntile * 16 + lm;
  float bias = fcb[n];
#pragma unroll
  for (int r = 0; r < 8; ++r) {
    int ms = mtile * 16 + r + 8 * half;
    ac[(long)ms * Hh + n] = (f16)(acc[r] + bias);
  }
}

// ============================================================
// Persistent autoregressive LSTM (T=512 steps, grid barrier, 2-way K-split)
// ============================================================
struct LstmArgs {
  const f16 *acoustic, *wih0, *whh0, *wih1, *whh1, *wpost, *zrow;
  const float *bih0, *bhh0, *bih1, *bhh1, *postb, *emb;
  f16 *h1, *h2, *prev;
  float *c1, *c2, *z1, *z2, *z3;   // z1/z2: [2][8][3072], z3: [2][8][440]
  float *out;
  unsigned *bar;   // [0]=counter [1]=generation
};

DEV void grid_barrier(unsigned* bar, unsigned nb) {
  __syncthreads();
  if (threadIdx.x == 0) {
    __threadfence();
    unsigned g = __hip_atomic_load(&bar[1], __ATOMIC_RELAXED, __HIP_MEMORY_SCOPE_AGENT);
    unsigned arrived = __hip_atomic_fetch_add(&bar[0], 1u, __ATOMIC_ACQ_REL, __HIP_MEMORY_SCOPE_AGENT);
    if (arrived == nb - 1) {
      __hip_atomic_store(&bar[0], 0u, __ATOMIC_RELAXED, __HIP_MEMORY_SCOPE_AGENT);
      __hip_atomic_fetch_add(&bar[1], 1u, __ATOMIC_ACQ_REL, __HIP_MEMORY_SCOPE_AGENT);
    } else {
      while (__hip_atomic_load(&bar[1], __ATOMIC_ACQUIRE, __HIP_MEMORY_SCOPE_AGENT) == g)
        __builtin_amdgcn_s_sleep(2);
    }
  }
  __syncthreads();
}

__global__ void k_lstm_init(f16* h1, f16* h2, f16* prev, float* c1, float* c2,
                            f16* zrow, unsigned* bar) {
  int i = blockIdx.x * blockDim.x + threadIdx.x;
  if (i < Bv * Hh) { h1[i] = (f16)0.0f; h2[i] = (f16)0.0f; c1[i] = 0.0f; c2[i] = 0.0f; }
  if (i < Bv * PREVN) prev[i] = (f16)0.0f;
  if (i < 1024) zrow[i] = (f16)0.0f;
  if (i < 2) bar[i] = 0u;
}

__global__ void k_lstm(LstmArgs A) {
  const int tid  = blockIdx.x * blockDim.x + threadIdx.x;
  const int wid  = tid >> 5;           // 0..383
  const int lane = threadIdx.x & 31;
  const int half = lane >> 4;
  const int lm   = lane & 15;
  const int wseg = wid % 192;          // N-tile
  const int ks   = wid / 192;          // K-split half (0/1)
  const bool rowok = (lm < Bv);

  // zero-row trick: padding rows (lm>=8) read from a zeroed 2KB (1024 f16) scratch row,
  // so all A-fragment loads are unconditional (address select only)
  const f16* h1r = rowok ? (A.h1 + lm * Hh) : A.zrow;
  const f16* h2r = rowok ? (A.h2 + lm * Hh) : A.zrow;
  const f16* pvr = rowok ? (A.prev + lm * PREVN - Hh) : A.zrow;  // +kg, kg in [768,944)

  for (int t = 0; t < Tn; ++t) {
    // ---- phase A: z1[ks] = partial( xt @ w_ih0^T + h1 @ w_hh0^T ) ----
    {
      int n0 = wseg * 16;
      const f16* acr = rowok ? (A.acoustic + ((long)(lm * Tn + t)) * Hh) : A.zrow;
      auto a_x = [&](int kg) -> v8h { return ld8((kg < Hh ? acr : pvr) + kg); };
      auto a_h = [&](int kg) -> v8h { return ld8(h1r + kg); };
      v8f acc = {};
      if (ks == 0) {
        acc = gemm_rng(acc, a_x, A.wih0, K1, n0, G4, 0, 480);
        acc = gemm_rng(acc, a_h, A.whh0, Hh, n0, G4, 0, 384);
      } else {
        acc = gemm_rng(acc, a_x, A.wih0, K1, n0, G4, 480, K1);
        acc = gemm_rng(acc, a_h, A.whh0, Hh, n0, G4, 384, Hh);
      }
      if (half == 0) {
#pragma unroll
        for (int r = 0; r < 8; ++r) A.z1[(ks * 8 + r) * G4 + n0 + lm] = acc[r];
      }
    }
    grid_barrier(A.bar, NBLK);

    // ---- phase B: LSTM1 cell update (6144 lanes; sums both K-halves) ----
    if (tid < Bv * Hh) {
      int b = tid / Hh, j = tid - b * Hh;
      const float* za = A.z1 + b * G4;
      const float* zb = A.z1 + (8 + b) * G4;
      float zi = za[j]          + zb[j]          + A.bih0[j]          + A.bhh0[j];
      float zf = za[Hh + j]     + zb[Hh + j]     + A.bih0[Hh + j]     + A.bhh0[Hh + j];
      float zg = za[2 * Hh + j] + zb[2 * Hh + j] + A.bih0[2 * Hh + j] + A.bhh0[2 * Hh + j];
      float zo = za[3 * Hh + j] + zb[3 * Hh + j] + A.bih0[3 * Hh + j] + A.bhh0[3 * Hh + j];
      float c = sigmf(zf) * A.c1[b * Hh + j] + sigmf(zi) * tanhf(zg);
      float h = sigmf(zo) * tanhf(c);
      A.c1[b * Hh + j] = c;
      A.h1[b * Hh + j] = (f16)h;
    }
    grid_barrier(A.bar, NBLK);

    // ---- phase C: z2[ks] = partial( h1 @ w_ih1^T + h2 @ w_hh1^T ) ----
    {
      int n0 = wseg * 16;
      auto a_h1 = [&](int kg) -> v8h { return ld8(h1r + kg); };
      auto a_h2 = [&](int kg) -> v8h { return ld8(h2r + kg); };
      v8f acc = {};
      int kb = ks ? 384 : 0, ke = ks ? Hh : 384;
      acc = gemm_rng(acc, a_h1, A.wih1, Hh, n0, G4, kb, ke);
      acc = gemm_rng(acc, a_h2, A.whh1, Hh, n0, G4, kb, ke);
      if (half == 0) {
#pragma unroll
        for (int r = 0; r < 8; ++r) A.z2[(ks * 8 + r) * G4 + n0 + lm] = acc[r];
      }
    }
    grid_barrier(A.bar, NBLK);

    // ---- phase D: LSTM2 cell update ----
    if (tid < Bv * Hh) {
      int b = tid / Hh, j = tid - b * Hh;
      const float* za = A.z2 + b * G4;
      const float* zb = A.z2 + (8 + b) * G4;
      float zi = za[j]          + zb[j]          + A.bih1[j]          + A.bhh1[j];
      float zf = za[Hh + j]     + zb[Hh + j]     + A.bih1[Hh + j]     + A.bhh1[Hh + j];
      float zg = za[2 * Hh + j] + zb[2 * Hh + j] + A.bih1[2 * Hh + j] + A.bhh1[2 * Hh + j];
      float zo = za[3 * Hh + j] + zb[3 * Hh + j] + A.bih1[3 * Hh + j] + A.bhh1[3 * Hh + j];
      float c = sigmf(zf) * A.c2[b * Hh + j] + sigmf(zi) * tanhf(zg);
      float h = sigmf(zo) * tanhf(c);
      A.c2[b * Hh + j] = c;
      A.h2[b * Hh + j] = (f16)h;
    }
    grid_barrier(A.bar, NBLK);

    // ---- phase E: logits z3[ks] = partial( h2 @ post_w^T )  (2x28 tiles) ----
    if (wid < 56) {
      int n0 = (wid % 28) * 16;
      int es = wid / 28;
      auto a_h2 = [&](int kg) -> v8h { return ld8(h2r + kg); };
      v8f acc = {};
      acc = gemm_rng(acc, a_h2, A.wpost, Hh, n0, NPOST, es ? 384 : 0, es ? Hh : 384);
      int n = n0 + lm;
      if (half == 0 && n < NPOST) {
#pragma unroll
        for (int r = 0; r < 8; ++r) A.z3[(es * 8 + r) * NPOST + n] = acc[r];
      }
    }
    grid_barrier(A.bar, NBLK);

    // ---- phase F: argmax over 5 classes, emit idx, feed back embedding ----
    if (tid < Bv * NP) {
      int b = tid / NP, p = tid - b * NP;
      const float* za = A.z3 + b * NPOST + p * NC;
      const float* zb = A.z3 + (8 + b) * NPOST + p * NC;
      float best = -3.4e38f; int bi = 0;
#pragma unroll
      for (int c = 0; c < NC; ++c) {
        float v = za[c] + zb[c] + A.postb[p * NC + c];
        if (v > best) { best = v; bi = c; }
      }
      A.out[((long)(b * Tn + t)) * NP + p] = (float)bi;
      A.prev[b * PREVN + p * 2 + 0] = (f16)A.emb[bi * 2 + 0];
      A.prev[b * PREVN + p * 2 + 1] = (f16)A.emb[bi * 2 + 1];
    }
    grid_barrier(A.bar, NBLK);
  }
}

// ============================================================
// host launcher
// ============================================================
extern "C" void kernel_launch(void* const* d_in, const int* in_sizes, int n_in,
                              void* d_out, int out_size, void* d_ws, size_t ws_size,
                              hipStream_t stream) {
  (void)in_sizes; (void)n_in; (void)out_size; (void)ws_size;

  const float* mel   = (const float*)d_in[0];
  const float* c1w   = (const float*)d_in[1];
  const float* c1b   = (const float*)d_in[2];
  const float* g1    = (const float*)d_in[3];
  const float* b1    = (const float*)d_in[4];
  const float* m1    = (const float*)d_in[5];
  const float* v1    = (const float*)d_in[6];
  const float* c2w   = (const float*)d_in[7];
  const float* c2b   = (const float*)d_in[8];
  const float* g2    = (const float*)d_in[9];
  const float* b2    = (const float*)d_in[10];
  const float* m2    = (const float*)d_in[11];
  const float* v2    = (const float*)d_in[12];
  const float* c3w   = (const float*)d_in[13];
  const float* c3b   = (const float*)d_in[14];
  const float* g3    = (const float*)d_in[15];
  const float* b3    = (const float*)d_in[16];
  const float* m3    = (const float*)d_in[17];
  const float* v3    = (const float*)d_in[18];
  const float* fcw   = (const float*)d_in[19];
  const float* fcb   = (const float*)d_in[20];
  const float* wih0  = (const float*)d_in[21];
  const float* whh0  = (const float*)d_in[22];
  const float* bih0  = (const float*)d_in[23];
  const float* bhh0  = (const float*)d_in[24];
  const float* wih1  = (const float*)d_in[25];
  const float* whh1  = (const float*)d_in[26];
  const float* bih1  = (const float*)d_in[27];
  const float* bhh1  = (const float*)d_in[28];
  const float* postw = (const float*)d_in[29];
  const float* postb = (const float*)d_in[30];
  const float* emb   = (const float*)d_in[31];

  // ---- workspace bump allocator (256B aligned) ----
  char* ws = (char*)d_ws;
  size_t off = 0;
  auto alloc = [&](size_t bytes) -> char* {
    off = (off + 255) & ~(size_t)255;
    char* p = ws + off;
    off += bytes;
    return p;
  };

  f16* wfc16   = (f16*)alloc((size_t)Hh * FCK * 2);
  f16* wih016  = (f16*)alloc((size_t)G4 * K1 * 2);
  f16* whh016  = (f16*)alloc((size_t)G4 * Hh * 2);
  f16* wih116  = (f16*)alloc((size_t)G4 * Hh * 2);
  f16* whh116  = (f16*)alloc((size_t)G4 * Hh * 2);
  f16* wpost16 = (f16*)alloc((size_t)NPOST * Hh * 2);
  f16* w216    = (f16*)alloc((size_t)C1 * C1 * 9 * 2);
  f16* w316    = (f16*)alloc((size_t)C3v * C1 * 9 * 2);
  f16* ac16    = (f16*)alloc((size_t)Bv * Tn * Hh * 2);
  f16* h1b     = (f16*)alloc((size_t)Bv * Hh * 2);
  f16* h2b     = (f16*)alloc((size_t)Bv * Hh * 2);
  f16* prevb   = (f16*)alloc((size_t)Bv * PREVN * 2);
  f16* zrow    = (f16*)alloc((size_t)1024 * 2);
  float* c1f   = (float*)alloc((size_t)Bv * Hh * 4);
  float* c2f   = (float*)alloc((size_t)Bv * Hh * 4);
  float* z1    = (float*)alloc((size_t)2 * Bv * G4 * 4);
  float* z2    = (float*)alloc((size_t)2 * Bv * G4 * 4);
  float* z3    = (float*)alloc((size_t)2 * Bv * NPOST * 4);
  unsigned* bar = (unsigned*)alloc(256);
  size_t n_x1p  = (size_t)Bv * P1T * P1R * C1;      // 45.6M elems (padded)
  size_t n_y2   = (size_t)Bv * Tn * Fm * C1;        // 45.0M
  size_t n_x2pp = (size_t)Bv * P2T * P2R * C1;      // 22.9M (padded)
  f16* x1p  = (f16*)alloc(n_x1p * 2);
  f16* y2   = (f16*)alloc(n_y2 * 2);
  f16* x2pp = (f16*)alloc(n_x2pp * 2);
  f16* y3   = x1p;  // reuse: 8*512*114*96 = 44.8M <= 45.6M
  f16* x3p  = y2;   // reuse: 8*512*57*96 = 22.4M <= 45.0M

  // ---- 0. zero padded activation buffers (halo => SAME padding) ----
  k_zero8<<<(int)((n_x1p / 8 + 255) / 256), 256, 0, stream>>>(x1p, (int)(n_x1p / 8));
  k_zero8<<<(int)((n_x2pp / 8 + 255) / 256), 256, 0, stream>>>(x2pp, (int)(n_x2pp / 8));

  // ---- 1. weight conversion / permutation ----
  struct { const float* s; f16* d; int n; } cv[5] = {
      {wih0,  wih016,  G4 * K1},
      {whh0,  whh016,  G4 * Hh},
      {wih1,  wih116,  G4 * Hh},
      {whh1,  whh116,  G4 * Hh},
      {postw, wpost16, NPOST * Hh}};
  for (int i = 0; i < 5; ++i)
    k_f32_to_f16<<<(cv[i].n + 255) / 256, 256, 0, stream>>>(cv[i].s, cv[i].d, cv[i].n);
  k_cvt_conv<<<(C1 * C1 * 9 + 255) / 256, 256, 0, stream>>>(c2w, w216, C1, C1);
  k_cvt_conv<<<(C3v * C1 * 9 + 255) / 256, 256, 0, stream>>>(c3w, w316, C3v, C1);
  k_cvt_fc<<<(Hh * FCK + 255) / 256, 256, 0, stream>>>(fcw, wfc16, Hh * FCK);

  // ---- 2. conv stack ----
  {
    int n = Bv * Tn * Fm * (C1 / 8);
    k_conv1_cl<<<(n + 255) / 256, 256, 0, stream>>>(mel, c1w, c1b, g1, b1, m1, v1, x1p, n);
  }
  {
    int tiles = (Bv * Tn * Fm / 16) * 3;      // 175872
    k_conv2_wmma<<<tiles / 8, 256, 0, stream>>>(x1p, w216, c2b, g2, b2, m2, v2, y2);
  }
  {
    int n = Bv * Tn * F2 * (C1 / 8);
    k_pool_pad<<<(n + 255) / 256, 256, 0, stream>>>(y2, x2pp, n);
  }
  {
    int tiles = (Bv * Tn * F2 / 16) * 6;      // 175104
    k_conv3_wmma<<<tiles / 8, 256, 0, stream>>>(x2pp, w316, c3b, g3, b3, m3, v3, y3);
  }
  {
    int n = Bv * Tn * F4 * (C3v / 8);
    k_pool_cl<<<(n + 255) / 256, 256, 0, stream>>>(y3, x3p, n, F2, F4, C3v);
  }
  {
    int tiles = (Bv * Tn / 16) * 48;          // 12288
    k_fc_wmma<<<tiles / 8, 256, 0, stream>>>(x3p, wfc16, fcb, ac16);
  }

  // ---- 3. persistent autoregressive LSTM ----
  k_lstm_init<<<(Bv * Hh + 255) / 256, 256, 0, stream>>>(h1b, h2b, prevb, c1f, c2f, zrow, bar);

  LstmArgs A;
  A.acoustic = ac16; A.wih0 = wih016; A.whh0 = whh016;
  A.wih1 = wih116;   A.whh1 = whh116; A.wpost = wpost16; A.zrow = zrow;
  A.bih0 = bih0; A.bhh0 = bhh0; A.bih1 = bih1; A.bhh1 = bhh1;
  A.postb = postb; A.emb = emb;
  A.h1 = h1b; A.h2 = h2b; A.prev = prevb;
  A.c1 = c1f; A.c2 = c2f; A.z1 = z1; A.z2 = z2; A.z3 = z3;
  A.out = (float*)d_out; A.bar = bar;
  k_lstm<<<NBLK, 256, 0, stream>>>(A);
}
